// GraphTransformer_12567074308297
// MI455X (gfx1250) — compile-verified, matrix operand order checked by reading, not run on hardware
//
#include <hip/hip_runtime.h>
#include <math.h>

#define HID   128
#define HEADS 8
#define DH    16
#define NLAYER 4
#define NCLS  10
#define KPE   16

typedef __attribute__((ext_vector_type(16))) __bf16 v16bf;
typedef __attribute__((ext_vector_type(2)))  __bf16 v2bf;
typedef __attribute__((ext_vector_type(8)))  float  v8f;

enum { ACT_NONE = 0, ACT_RELU = 1, ACT_GELU = 2 };

__device__ __forceinline__ float gelu_exact(float x) {
  return 0.5f * x * (1.0f + erff(x * 0.70710678118654752440f));
}

// ---------------------------------------------------------------------------
// WMMA GEMM: Y[nrows,128] = act( X[nrows,KDIM] @ W[KDIM,128] + bias )
// 256 threads = 8 waves; block computes 128 rows; wave computes a 16x128 strip
// as 8 column tiles accumulated over KDIM/32 WMMA K-steps.
//  * All A global loads issued before the LDS barrier (overlap W staging).
//  * Per-K-step dedicated A fragments: no WMMA->VALU WAR hazard, no reload
//    serialization.
//  * W (fp32) -> bf16, staged transposed in LDS [n][k], pitch 136 (4-bank
//    stride across n -> conflict-free ds_load_b128 fragment reads).
// ---------------------------------------------------------------------------
template <int KDIM>
__global__ __launch_bounds__(256)
void gemm_bf16_wmma(const float* __restrict__ X, const float* __restrict__ W,
                    const float* __restrict__ bias, float* __restrict__ Y,
                    int nrows, int act)
{
  constexpr int NKC = KDIM >> 5;      // WMMA K-steps (32 each)
  __shared__ __bf16 sW[128][136];     // [n_out][k]

  const int tid  = threadIdx.x;
  const int wave = tid >> 5;
  const int lane = tid & 31;
  const int l15  = lane & 15;
  const int half = lane >> 4;

  // ---- Issue all A loads first (in flight during W staging + barrier) ----
  // 16-bit A 16x32 layout: M = lane%16; lanes<16 take K {0..7,16..23}+kc*32,
  // lanes>=16 take K {8..15,24..31}+kc*32  ->  float4 indices q0,q0+1,q0+4,q0+5.
  const int m_frag = blockIdx.x * 128 + wave * 16 + l15;
  const int m_ld   = (m_frag < nrows) ? m_frag : (nrows - 1);   // clamp: no predication
  const float4* __restrict__ xr4 = (const float4*)(X + (size_t)m_ld * KDIM);

  float4 af[NKC * 4];
  #pragma unroll
  for (int kc = 0; kc < NKC; ++kc) {
    const int q0 = kc * 8 + half * 2;
    af[kc * 4 + 0] = xr4[q0];
    af[kc * 4 + 1] = xr4[q0 + 1];
    af[kc * 4 + 2] = xr4[q0 + 4];
    af[kc * 4 + 3] = xr4[q0 + 5];
  }

  // ---- Stage W transposed: K-adjacent pair -> one aligned b32 DS store ----
  for (int idx = tid; idx < (KDIM / 2) * 128; idx += 256) {
    const int kh = idx >> 7;          // k/2
    const int n  = idx & 127;
    const int k  = kh * 2;
    v2bf pk;
    pk[0] = (__bf16)W[(size_t)k * 128 + n];
    pk[1] = (__bf16)W[(size_t)(k + 1) * 128 + n];
    *(v2bf*)&sW[n][k] = pk;           // 4B-aligned: pitch 272B, k even
  }

  // ---- Convert A to per-step bf16 fragments ----
  v16bf afrag[NKC];
  #pragma unroll
  for (int kc = 0; kc < NKC; ++kc) {
    #pragma unroll
    for (int j = 0; j < 4; ++j) {
      const float4 f = af[kc * 4 + j];
      afrag[kc][j * 4 + 0] = (__bf16)f.x;
      afrag[kc][j * 4 + 1] = (__bf16)f.y;
      afrag[kc][j * 4 + 2] = (__bf16)f.z;
      afrag[kc][j * 4 + 3] = (__bf16)f.w;
    }
  }
  __syncthreads();

  v8f acc[8];
  #pragma unroll
  for (int t = 0; t < 8; ++t) {
    #pragma unroll
    for (int i = 0; i < 8; ++i) acc[t][i] = 0.0f;
  }

  #pragma unroll
  for (int kc = 0; kc < NKC; ++kc) {
    const int kbase = kc * 32 + half * 8;
    #pragma unroll
    for (int tn = 0; tn < 8; ++tn) {
      const int n = tn * 16 + l15;              // lane -> output column
      v16bf b;
      #pragma unroll
      for (int e = 0; e < 8; ++e) {
        b[e]     = sW[n][kbase + e];
        b[e + 8] = sW[n][kbase + 16 + e];
      }
      acc[tn] = __builtin_amdgcn_wmma_f32_16x16x32_bf16(
          false, afrag[kc], false, b, (short)0, acc[tn], false, false);
    }
  }

  // Epilogue. f32 C/D layout: VGPR i, lanes<16 -> row i, lanes>=16 -> row i+8.
  const int r_base = blockIdx.x * 128 + wave * 16 + half * 8;
  #pragma unroll
  for (int i = 0; i < 8; ++i) {
    const int r = r_base + i;
    if (r >= nrows) continue;
    #pragma unroll
    for (int tn = 0; tn < 8; ++tn) {
      const int cidx = tn * 16 + l15;
      float v = acc[tn][i];
      if (bias) v += bias[cidx];
      if (act == ACT_RELU)      v = fmaxf(v, 0.0f);
      else if (act == ACT_GELU) v = gelu_exact(v);
      Y[(size_t)r * 128 + cidx] = v;
    }
  }
}

// ---------------------------------------------------------------------------
// PE phi1: S[n,c] = sum_k relu(ev0[n,k]*W1[0,c] + ev1[n,k]*W1[1,c])
// ---------------------------------------------------------------------------
__global__ __launch_bounds__(128)
void pe_phi1_kernel(const float* __restrict__ eigvecs, const float* __restrict__ eigvals,
                    const float* __restrict__ eps_pe, const float* __restrict__ phi_W1,
                    float* __restrict__ S)
{
  __shared__ float se0[KPE], se1[KPE];
  const int n = blockIdx.x, c = threadIdx.x;
  if (c < KPE) {
    float v0 = eigvecs[(size_t)n * KPE + c];
    float v1 = eigvals[(size_t)n * KPE + c] + eps_pe[c];
    se0[c] = (v0 == v0) ? v0 : 0.0f;  // nan -> 0
    se1[c] = (v1 == v1) ? v1 : 0.0f;
  }
  __syncthreads();
  const float w0 = phi_W1[c], w1 = phi_W1[HID + c];
  float acc = 0.0f;
  #pragma unroll
  for (int kk = 0; kk < KPE; ++kk)
    acc += fmaxf(se0[kk] * w0 + se1[kk] * w1, 0.0f);
  S[(size_t)n * HID + c] = acc;
}

// ---------------------------------------------------------------------------
// Row LayerNorm (128 cols). flags bit0 = gelu(input) first, bit1 = Y += result.
// Optional addsrc added after LN affine.
// ---------------------------------------------------------------------------
__global__ __launch_bounds__(128)
void row_ln_kernel(const float* __restrict__ X, const float* __restrict__ g,
                   const float* __restrict__ b, const float* __restrict__ addsrc,
                   float* __restrict__ Y, int flags)
{
  __shared__ float red[128];
  const int n = blockIdx.x, c = threadIdx.x;
  float v = X[(size_t)n * HID + c];
  if (flags & 1) v = gelu_exact(v);

  red[c] = v; __syncthreads();
  for (int s = 64; s > 0; s >>= 1) { if (c < s) red[c] += red[c + s]; __syncthreads(); }
  const float mean = red[0] * (1.0f / 128.0f); __syncthreads();

  const float d = v - mean;
  red[c] = d * d; __syncthreads();
  for (int s = 64; s > 0; s >>= 1) { if (c < s) red[c] += red[c + s]; __syncthreads(); }
  const float var = red[0] * (1.0f / 128.0f);

  float r = d * rsqrtf(var + 1e-5f) * g[c] + b[c];
  if (addsrc) r += addsrc[(size_t)n * HID + c];
  if (flags & 2) Y[(size_t)n * HID + c] += r;
  else           Y[(size_t)n * HID + c]  = r;
}

// ---------------------------------------------------------------------------
// Edge-softmax attention (TransformerConv), 3 atomic passes.
// ---------------------------------------------------------------------------
__device__ __forceinline__ void atomicMaxFloat(float* addr, float val) {
  if (!(val < 0.0f)) atomicMax((int*)addr, __float_as_int(val));
  else               atomicMin((unsigned int*)addr, __float_as_uint(val));
}

__global__ void attn_init_kernel(float* smax, float* denom, int count) {
  int i = blockIdx.x * blockDim.x + threadIdx.x;
  if (i < count) { smax[i] = __int_as_float(0xff800000); denom[i] = 0.0f; }
}

__global__ void attn_score_kernel(const int* __restrict__ src, const int* __restrict__ dst,
                                  const float* __restrict__ q, const float* __restrict__ k,
                                  float* __restrict__ score, float* __restrict__ smax, int nE)
{
  int i = blockIdx.x * blockDim.x + threadIdx.x;
  if (i >= nE * HEADS) return;
  const int e = i >> 3, hh = i & 7;
  const int s = src[e], d = dst[e];
  const float* qp = q + (size_t)d * HID + hh * DH;
  const float* kp = k + (size_t)s * HID + hh * DH;
  float acc = 0.0f;
  #pragma unroll
  for (int j = 0; j < DH; ++j) acc += qp[j] * kp[j];
  acc *= 0.25f;                                   // 1/sqrt(16)
  score[i] = acc;
  atomicMaxFloat(&smax[d * HEADS + hh], acc);
}

__global__ void attn_exp_kernel(const int* __restrict__ dst, float* __restrict__ score,
                                const float* __restrict__ smax, float* __restrict__ denom, int nE)
{
  int i = blockIdx.x * blockDim.x + threadIdx.x;
  if (i >= nE * HEADS) return;
  const int e = i >> 3, hh = i & 7;
  const int d = dst[e];
  const float ex = expf(score[i] - smax[d * HEADS + hh]);
  score[i] = ex;
  atomicAdd(&denom[d * HEADS + hh], ex);
}

__global__ void attn_apply_kernel(const int* __restrict__ src, const int* __restrict__ dst,
                                  const float* __restrict__ score, const float* __restrict__ denom,
                                  const float* __restrict__ v, float* __restrict__ out, int nE)
{
  int i = blockIdx.x * blockDim.x + threadIdx.x;
  if (i >= nE * HEADS) return;
  const int e = i >> 3, hh = i & 7;
  const int s = src[e], d = dst[e];
  const float alpha = score[i] / denom[d * HEADS + hh];
  const float* vp = v   + (size_t)s * HID + hh * DH;
  float*       op = out + (size_t)d * HID + hh * DH;
  #pragma unroll
  for (int j = 0; j < DH; ++j) atomicAdd(&op[j], vp[j] * alpha);
}

// ---------------------------------------------------------------------------
// Head: out[n, 0..9] = Z[n,:] @ W2[128,10] + b2
// ---------------------------------------------------------------------------
__global__ void head_out_kernel(const float* __restrict__ Z, const float* __restrict__ W2,
                                const float* __restrict__ b2, float* __restrict__ out, int nrows)
{
  int n = blockIdx.x * blockDim.x + threadIdx.x;
  if (n >= nrows) return;
  float acc[NCLS];
  #pragma unroll
  for (int c = 0; c < NCLS; ++c) acc[c] = b2[c];
  const float* z = Z + (size_t)n * HID;
  for (int k = 0; k < HID; ++k) {
    const float zv = z[k];
    #pragma unroll
    for (int c = 0; c < NCLS; ++c) acc[c] += zv * W2[k * NCLS + c];
  }
  #pragma unroll
  for (int c = 0; c < NCLS; ++c) out[(size_t)n * NCLS + c] = acc[c];
}

// ---------------------------------------------------------------------------
extern "C" void kernel_launch(void* const* d_in, const int* in_sizes, int n_in,
                              void* d_out, int out_size, void* d_ws, size_t ws_size,
                              hipStream_t stream)
{
  (void)n_in; (void)out_size; (void)ws_size;

  const int N = in_sizes[0] / 64;     // x is [N,64]
  const int E = in_sizes[1] / 2;      // edge_index is [2,E]

  const float* x       = (const float*)d_in[0];
  const int*   eidx    = (const int*)d_in[1];
  const float* eigvecs = (const float*)d_in[2];
  const float* eigvals = (const float*)d_in[3];
  int p = 4;
  const float* W_enc   = (const float*)d_in[p++];
  const float* b_enc   = (const float*)d_in[p++];
  const float* phi_W1  = (const float*)d_in[p++];
  const float* phi_W2  = (const float*)d_in[p++];
  const float* rho_W1  = (const float*)d_in[p++];
  const float* rho_b1  = (const float*)d_in[p++];
  const float* rho_W2  = (const float*)d_in[p++];
  const float* rho_b2  = (const float*)d_in[p++];
  const float* eps_pe  = (const float*)d_in[p++];
  const float* in_g    = (const float*)d_in[p++];
  const float* in_b    = (const float*)d_in[p++];
  const float* Wq      = (const float*)d_in[p++];
  const float* bq      = (const float*)d_in[p++];
  const float* Wk      = (const float*)d_in[p++];
  const float* bk      = (const float*)d_in[p++];
  const float* Wv      = (const float*)d_in[p++];
  const float* bv      = (const float*)d_in[p++];
  const float* Wskip   = (const float*)d_in[p++];
  const float* bskip   = (const float*)d_in[p++];
  const float* ln_g    = (const float*)d_in[p++];
  const float* ln_b    = (const float*)d_in[p++];
  const float* head_W1 = (const float*)d_in[p++];
  const float* head_b1 = (const float*)d_in[p++];
  const float* head_g  = (const float*)d_in[p++];
  const float* head_bn = (const float*)d_in[p++];
  const float* head_W2 = (const float*)d_in[p++];
  const float* head_b2 = (const float*)d_in[p++];

  const int* src = eidx;          // edge_index[0]
  const int* dst = eidx + E;      // edge_index[1]

  // Workspace carve-out
  const size_t NB = (size_t)N * HID;
  float* B0 = (float*)d_ws;       // h (persistent hidden state)
  float* B1 = B0 + NB;
  float* B2 = B1 + NB;
  float* B3 = B2 + NB;
  float* B4 = B3 + NB;
  float* SC = B4 + NB;                  // edge scores / exp  [E*H]
  float* SM = SC + (size_t)E * HEADS;   // segment max [N*H]
  float* DN = SM + (size_t)N * HEADS;   // segment sum [N*H]

  const dim3 gblk((N + 127) / 128), gthr(256);
  const dim3 rowg(N), rowt(128);
  const int  EH = E * HEADS;
  const dim3 eg((EH + 255) / 256), et(256);
  const int  NH = N * HEADS;

  // ---- Encoder + Laplacian PE ----
  gemm_bf16_wmma<64> <<<gblk, gthr, 0, stream>>>(x, W_enc, b_enc, B1, N, ACT_NONE);
  pe_phi1_kernel<<<rowg, rowt, 0, stream>>>(eigvecs, eigvals, eps_pe, phi_W1, B2);
  gemm_bf16_wmma<128><<<gblk, gthr, 0, stream>>>(B2, phi_W2, nullptr, B3, N, ACT_NONE); // t = s @ phi_W2
  gemm_bf16_wmma<128><<<gblk, gthr, 0, stream>>>(B3, rho_W1, rho_b1, B2, N, ACT_RELU);
  gemm_bf16_wmma<128><<<gblk, gthr, 0, stream>>>(B2, rho_W2, rho_b2, B3, N, ACT_NONE);  // pe
  // h = LN(enc)*in_g+in_b + pe
  row_ln_kernel<<<rowg, rowt, 0, stream>>>(B1, in_g, in_b, B3, B0, 0);

  // ---- Transformer layers ----
  for (int l = 0; l < NLAYER; ++l) {
    const size_t wo = (size_t)l * HID * HID, bo = (size_t)l * HID;
    gemm_bf16_wmma<128><<<gblk, gthr, 0, stream>>>(B0, Wq + wo, bq + bo, B1, N, ACT_NONE);   // q
    gemm_bf16_wmma<128><<<gblk, gthr, 0, stream>>>(B0, Wk + wo, bk + bo, B2, N, ACT_NONE);   // k
    gemm_bf16_wmma<128><<<gblk, gthr, 0, stream>>>(B0, Wv + wo, bv + bo, B3, N, ACT_NONE);   // v
    gemm_bf16_wmma<128><<<gblk, gthr, 0, stream>>>(B0, Wskip + wo, bskip + bo, B4, N, ACT_NONE); // skip (attn accumulates into it)

    attn_init_kernel <<<(NH + 255) / 256, 256, 0, stream>>>(SM, DN, NH);
    attn_score_kernel<<<eg, et, 0, stream>>>(src, dst, B1, B2, SC, SM, E);
    attn_exp_kernel  <<<eg, et, 0, stream>>>(dst, SC, SM, DN, E);
    attn_apply_kernel<<<eg, et, 0, stream>>>(src, dst, SC, DN, B3, B4, E);

    // h += LN(gelu(attn_out + skip)) * ln_g[l] + ln_b[l]
    row_ln_kernel<<<rowg, rowt, 0, stream>>>(B4, ln_g + bo, ln_b + bo, nullptr, B0, 1 | 2);
  }

  // ---- Head ----
  gemm_bf16_wmma<128><<<gblk, gthr, 0, stream>>>(B0, head_W1, head_b1, B1, N, ACT_GELU);
  row_ln_kernel <<<rowg, rowt, 0, stream>>>(B1, head_g, head_bn, nullptr, B2, 0);
  head_out_kernel<<<(N + 255) / 256, 256, 0, stream>>>(B2, head_W2, head_b2, (float*)d_out, N);
}
